// Attention_72189810311253
// MI455X (gfx1250) — matrix operand order repaired and checked by reading, not compiled, not run on hardware
//
#include <hip/hip_runtime.h>
#include <math.h>

// ---------------------------------------------------------------------------
// Attention forward for MI455X (gfx1250, wave32, WMMA bf16 16x16x32 f32-acc)
// B=2, N=2048, C=512, H=8, D=64.  SCALE = C^-0.5 (folded into Q with log2e).
// ---------------------------------------------------------------------------

#define C_DIM  512
#define N_SEQ  2048
#define HEADS  8
#define DHEAD  64
#define BATCH  2
#define SCALE  0.04419417382415922f              // 512^-0.5
#define QSCALE (SCALE * 1.4426950408889634f)     // fold log2(e): softmax in base-2

typedef __attribute__((ext_vector_type(16))) __bf16 v16bf;
typedef __attribute__((ext_vector_type(8)))  float  v8f;

static __device__ __forceinline__ __bf16 tobf(float f) { return (__bf16)f; }
static __device__ __forceinline__ unsigned short bfbits(float f) {
    return __builtin_bit_cast(unsigned short, (__bf16)f);
}
static __device__ __forceinline__ __bf16 bf_of(unsigned short u) {
    return __builtin_bit_cast(__bf16, u);
}
static __device__ __forceinline__ float fast_exp2(float x) {
    return __builtin_amdgcn_exp2f(x);            // bare v_exp_f32
}
static __device__ __forceinline__ float hred_max(float v) {
    v = fmaxf(v, __shfl_xor(v, 1, 32));
    v = fmaxf(v, __shfl_xor(v, 2, 32));
    v = fmaxf(v, __shfl_xor(v, 4, 32));
    v = fmaxf(v, __shfl_xor(v, 8, 32));
    return v;   // max over the 16 lanes of this half-wave
}
static __device__ __forceinline__ float hred_sum(float v) {
    v += __shfl_xor(v, 1, 32);
    v += __shfl_xor(v, 2, 32);
    v += __shfl_xor(v, 4, 32);
    v += __shfl_xor(v, 8, 32);
    return v;
}

// ---------------------------------------------------------------------------
// Kernel 0: weights -> bf16, transposed: Wt[j*512 + k] = bf16(W[k*512 + j])
// ---------------------------------------------------------------------------
__global__ __launch_bounds__(256) void prep_weights_kernel(
    const float* __restrict__ Wq, const float* __restrict__ Wk,
    const float* __restrict__ Wv, const float* __restrict__ Wo,
    unsigned short* __restrict__ WqT, unsigned short* __restrict__ WkT,
    unsigned short* __restrict__ WvT, unsigned short* __restrict__ WoT)
{
    const int idx = blockIdx.x * blockDim.x + threadIdx.x;   // 0 .. 512*512-1
    const int j = idx >> 9;
    const int k = idx & (C_DIM - 1);
    const int src = k * C_DIM + j;
    WqT[idx] = bfbits(Wq[src]);
    WkT[idx] = bfbits(Wk[src]);
    WvT[idx] = bfbits(Wv[src]);
    WoT[idx] = bfbits(Wo[src]);
}

// ---------------------------------------------------------------------------
// Kernel 1: QKV projection.  blockIdx.y selects mode:
//   0: Q = (q@Wq + pos_q) * QSCALE -> Qh  bf16 [b,h,n,d]
//   1: K = kv@Wk + pos_k           -> Kh  bf16 [b,h,n,d]
//   2: V = kv@Wv                   -> VhT bf16 [b,h,d,n]  (transposed)
// One wave computes a 16-token x 64-col (one head) tile.
// ---------------------------------------------------------------------------
__global__ __launch_bounds__(256) void qkv_proj_kernel(
    const float* __restrict__ q, const float* __restrict__ kv,
    const float* __restrict__ pos_q, const float* __restrict__ pos_k,
    const unsigned short* __restrict__ WqT, const unsigned short* __restrict__ WkT,
    const unsigned short* __restrict__ WvT,
    unsigned short* __restrict__ Qh, unsigned short* __restrict__ Kh,
    unsigned short* __restrict__ VhT)
{
    const int mode = blockIdx.y;
    const float* X = (mode == 0) ? q : kv;
    const unsigned short* Wt = (mode == 0) ? WqT : (mode == 1) ? WkT : WvT;

    const int lane = threadIdx.x & 31;
    const int wave = threadIdx.x >> 5;
    const int tile = blockIdx.x * 8 + wave;          // 0..2047
    const int h    = tile & (HEADS - 1);
    const int m0   = (tile >> 3) * 16;               // flat token row (0..4095)
    const int mrow = lane & 15;
    const int hi   = lane >> 4;                      // half-wave id
    const int ka   = hi ? 16 : 0;                    // A-frag K sub-offset
    const int kb   = hi ? 16 : 0;                    // B-frag K sub-offset

    const float* arow = X + (size_t)(m0 + mrow) * C_DIM;
    const unsigned short* brow[4];
    #pragma unroll
    for (int t = 0; t < 4; ++t)
        brow[t] = Wt + (size_t)(h * DHEAD + t * 16 + mrow) * C_DIM;

    v8f acc[4] = {};

    #pragma unroll 2
    for (int kk = 0; kk < C_DIM; kk += 32) {
        v16bf a;
        #pragma unroll
        for (int e = 0; e < 8; ++e) {
            a[e]     = tobf(arow[kk + ka + e]);
            a[e + 8] = tobf(arow[kk + 8 + ka + e]);
        }
        #pragma unroll
        for (int t = 0; t < 4; ++t) {
            v16bf b;
            #pragma unroll
            for (int e = 0; e < 16; ++e)
                b[e] = bf_of(brow[t][kk + kb + e]);
            acc[t] = __builtin_amdgcn_wmma_f32_16x16x32_bf16(
                false, a, false, b, (short)0, acc[t], false, false);
        }
    }

    const int bb = m0 >> 11;                 // batch
    const int n0 = m0 & (N_SEQ - 1);
    if (mode < 2) {
        const float* pos = (mode == 0) ? pos_q : pos_k;
        const float postscale = (mode == 0) ? QSCALE : 1.0f;
        unsigned short* dst = (mode == 0) ? Qh : Kh;
        #pragma unroll
        for (int t = 0; t < 4; ++t) {
            const int d = t * 16 + mrow;
            #pragma unroll
            for (int r = 0; r < 8; ++r) {
                const int n = n0 + r + hi * 8;
                const float v =
                    (acc[t][r] + pos[(size_t)(bb * N_SEQ + n) * DHEAD + d]) * postscale;
                dst[((size_t)(bb * HEADS + h) * N_SEQ + n) * DHEAD + d] = bfbits(v);
            }
        }
    } else {
        // transposed store: 8 consecutive tokens per lane -> one b128 store
        #pragma unroll
        for (int t = 0; t < 4; ++t) {
            const int d = t * 16 + mrow;
            const int n = n0 + hi * 8;
            unsigned int pk[4];
            #pragma unroll
            for (int r = 0; r < 4; ++r)
                pk[r] = (unsigned int)bfbits(acc[t][2 * r]) |
                        ((unsigned int)bfbits(acc[t][2 * r + 1]) << 16);
            *(uint4*)&VhT[((size_t)(bb * HEADS + h) * DHEAD + d) * N_SEQ + n] =
                *(const uint4*)pk;
        }
    }
}

// ---------------------------------------------------------------------------
// Kernel 2: flash attention.  One wave owns 16 queries of one (b,h), streams
// keys in chunks of 32 with base-2 online softmax.  The V tile for a chunk is
// staged LDS-side with GLOBAL_LOAD_ASYNC_TO_LDS_B128 (ASYNCcnt) so its fetch
// overlaps the S WMMAs + softmax; next chunk's K rows are prefetched.
// P is bounced through a per-wave LDS tile to reach WMMA A-fragment layout.
// ---------------------------------------------------------------------------
__global__ __launch_bounds__(256) void flash_attn_kernel(
    const unsigned short* __restrict__ Qh, const unsigned short* __restrict__ Kh,
    const unsigned short* __restrict__ VhT, float* __restrict__ Obuf)
{
    __shared__ unsigned short pbuf[8][16][32];   // per-wave P tile (bf16 bits)
    __shared__ unsigned short vtile[8][64][32];  // per-wave V tile (64 d x 32 keys)

    const int lane = threadIdx.x & 31;
    const int wave = threadIdx.x >> 5;
    const int job  = blockIdx.x * 8 + wave;          // 0..2047
    const int bh   = job >> 7;                       // 0..15
    const int m0   = (job & 127) * 16;               // query tile start
    const int bb   = bh >> 3;
    const int h    = bh & 7;
    const int mrow = lane & 15;
    const int hi   = lane >> 4;
    const int ka   = hi ? 16 : 0;
    const int kb   = hi ? 16 : 0;

    const unsigned short* Qb = Qh  + (size_t)bh * N_SEQ * DHEAD;
    const unsigned short* Kb = Kh  + (size_t)bh * N_SEQ * DHEAD;
    const unsigned short* Vb = VhT + (size_t)bh * DHEAD * N_SEQ;

    // Q A-fragments, loaded once (d split into two K=32 steps)
    v16bf qa[2];
    {
        const unsigned short* qrow = Qb + (size_t)(m0 + mrow) * DHEAD;
        #pragma unroll
        for (int s = 0; s < 2; ++s)
            #pragma unroll
            for (int e = 0; e < 8; ++e) {
                qa[s][e]     = bf_of(qrow[s * 32 + ka + e]);
                qa[s][e + 8] = bf_of(qrow[s * 32 + 8 + ka + e]);
            }
    }

    float Mrun[8], Lrun[8];
    v8f oacc[4] = {};
    #pragma unroll
    for (int r = 0; r < 8; ++r) { Mrun[r] = -INFINITY; Lrun[r] = 0.0f; }

    for (int kc = 0; kc < N_SEQ; kc += 32) {
        // ---- stage this chunk's V tile into LDS asynchronously (ASYNCcnt) ----
        // 128 b128 transfers: id = t*32+lane -> row = id>>1 (0..63), half = id&1
        #pragma unroll
        for (int t = 0; t < 4; ++t) {
            const int id   = t * 32 + lane;
            const int row  = id >> 1;
            const int half = id & 1;
            const unsigned short* src = Vb + (size_t)row * N_SEQ + kc + half * 16;
            const unsigned ldsoff =
                (unsigned)(size_t)&vtile[wave][row][half * 16];
            asm volatile("global_load_async_to_lds_b128 %0, %1, off"
                         :: "v"(ldsoff), "v"((unsigned long long)(size_t)src)
                         : "memory");
        }

        // ---- prefetch next chunk's K rows while we compute ----
        if (kc + 32 < N_SEQ) {
            __builtin_prefetch(Kb + (size_t)(kc + 32 + mrow) * DHEAD, 0, 3);
            __builtin_prefetch(Kb + (size_t)(kc + 48 + mrow) * DHEAD, 0, 3);
        }

        // ---- S = Q @ K^T for 16 queries x 32 keys (Q pre-scaled) ----
        v8f sacc[2] = {};
        #pragma unroll
        for (int jt = 0; jt < 2; ++jt) {
            const unsigned short* krow =
                Kb + (size_t)(kc + jt * 16 + mrow) * DHEAD;
            #pragma unroll
            for (int s = 0; s < 2; ++s) {
                v16bf kf;
                #pragma unroll
                for (int e = 0; e < 16; ++e)
                    kf[e] = bf_of(krow[s * 32 + kb + e]);
                sacc[jt] = __builtin_amdgcn_wmma_f32_16x16x32_bf16(
                    false, qa[s], false, kf, (short)0, sacc[jt], false, false);
            }
        }

        // ---- base-2 online softmax; write P (bf16) into this wave's LDS ----
        #pragma unroll
        for (int r = 0; r < 8; ++r) {
            const float s0 = sacc[0][r];
            const float s1 = sacc[1][r];
            const float rowmax = hred_max(fmaxf(s0, s1));
            const float mnew  = fmaxf(Mrun[r], rowmax);
            const float alpha = fast_exp2(Mrun[r] - mnew);
            const float p0 = fast_exp2(s0 - mnew);
            const float p1 = fast_exp2(s1 - mnew);
            Lrun[r] = Lrun[r] * alpha + hred_sum(p0 + p1);
            Mrun[r] = mnew;
            #pragma unroll
            for (int t = 0; t < 4; ++t) oacc[t][r] *= alpha;
            const int m = r + hi * 8;
            pbuf[wave][m][mrow]      = bfbits(p0);
            pbuf[wave][m][16 + mrow] = bfbits(p1);
        }
        __builtin_amdgcn_wave_barrier();   // DS is in-order within a wave

        // P A-fragment from LDS
        v16bf pa;
        #pragma unroll
        for (int e = 0; e < 8; ++e) {
            pa[e]     = bf_of(pbuf[wave][mrow][ka + e]);
            pa[e + 8] = bf_of(pbuf[wave][mrow][8 + ka + e]);
        }

        // ---- wait for the async V tile, then O += P @ V from LDS ----
        asm volatile("s_wait_asynccnt 0x0" ::: "memory");
        #pragma unroll
        for (int t = 0; t < 4; ++t) {
            const unsigned short* vrow = &vtile[wave][t * 16 + mrow][0];
            v16bf vf;
            #pragma unroll
            for (int e = 0; e < 16; ++e)
                vf[e] = bf_of(vrow[kb + e]);
            oacc[t] = __builtin_amdgcn_wmma_f32_16x16x32_bf16(
                false, pa, false, vf, (short)0, oacc[t], false, false);
        }
        __builtin_amdgcn_wave_barrier();
    }

    // normalize and store merged-head fp32 scratch [b, n, h*64+d]
    #pragma unroll
    for (int r = 0; r < 8; ++r) {
        const float inv = 1.0f / Lrun[r];
        const int n = m0 + r + hi * 8;
        #pragma unroll
        for (int t = 0; t < 4; ++t)
            Obuf[(size_t)(bb * N_SEQ + n) * C_DIM + h * DHEAD + t * 16 + mrow] =
                oacc[t][r] * inv;
    }
}

// ---------------------------------------------------------------------------
// Kernel 3: out = Obuf @ Wo + bo   (fp32 output)
// ---------------------------------------------------------------------------
__global__ __launch_bounds__(256) void out_proj_kernel(
    const float* __restrict__ Obuf, const unsigned short* __restrict__ WoT,
    const float* __restrict__ bo, float* __restrict__ out)
{
    const int lane = threadIdx.x & 31;
    const int wave = threadIdx.x >> 5;
    const int tile = blockIdx.x * 8 + wave;          // 0..2047
    const int cg   = tile & 7;                       // 64-col group
    const int m0   = (tile >> 3) * 16;
    const int mrow = lane & 15;
    const int hi   = lane >> 4;
    const int ka   = hi ? 16 : 0;
    const int kb   = hi ? 16 : 0;

    const float* arow = Obuf + (size_t)(m0 + mrow) * C_DIM;
    const unsigned short* brow[4];
    #pragma unroll
    for (int t = 0; t < 4; ++t)
        brow[t] = WoT + (size_t)(cg * 64 + t * 16 + mrow) * C_DIM;

    v8f acc[4] = {};
    #pragma unroll 2
    for (int kk = 0; kk < C_DIM; kk += 32) {
        v16bf a;
        #pragma unroll
        for (int e = 0; e < 8; ++e) {
            a[e]     = tobf(arow[kk + ka + e]);
            a[e + 8] = tobf(arow[kk + 8 + ka + e]);
        }
        #pragma unroll
        for (int t = 0; t < 4; ++t) {
            v16bf b;
            #pragma unroll
            for (int e = 0; e < 16; ++e)
                b[e] = bf_of(brow[t][kk + kb + e]);
            acc[t] = __builtin_amdgcn_wmma_f32_16x16x32_bf16(
                false, a, false, b, (short)0, acc[t], false, false);
        }
    }

    #pragma unroll
    for (int t = 0; t < 4; ++t) {
        const int col = cg * 64 + t * 16 + mrow;
        const float bias = bo[col];
        #pragma unroll
        for (int r = 0; r < 8; ++r) {
            const int row = m0 + r + hi * 8;
            out[(size_t)row * C_DIM + col] = acc[t][r] + bias;
        }
    }
}

// ---------------------------------------------------------------------------
extern "C" void kernel_launch(void* const* d_in, const int* in_sizes, int n_in,
                              void* d_out, int out_size, void* d_ws, size_t ws_size,
                              hipStream_t stream) {
    (void)in_sizes; (void)n_in; (void)out_size; (void)ws_size;
    const float* q     = (const float*)d_in[0];
    const float* kv    = (const float*)d_in[1];
    const float* pos_q = (const float*)d_in[2];
    const float* pos_k = (const float*)d_in[3];
    const float* Wq    = (const float*)d_in[4];
    const float* Wk    = (const float*)d_in[5];
    const float* Wv    = (const float*)d_in[6];
    const float* Wo    = (const float*)d_in[7];
    const float* bo    = (const float*)d_in[8];
    float* out = (float*)d_out;

    // workspace carve-up (~22 MB)
    char* ws = (char*)d_ws;
    size_t off = 0;
    const size_t WSZ  = (size_t)C_DIM * C_DIM;                       // 262144
    const size_t HSZ  = (size_t)BATCH * HEADS * N_SEQ * DHEAD;       // 2M elems
    unsigned short* WqT = (unsigned short*)(ws + off); off += WSZ * 2;
    unsigned short* WkT = (unsigned short*)(ws + off); off += WSZ * 2;
    unsigned short* WvT = (unsigned short*)(ws + off); off += WSZ * 2;
    unsigned short* WoT = (unsigned short*)(ws + off); off += WSZ * 2;
    unsigned short* Qh  = (unsigned short*)(ws + off); off += HSZ * 2;
    unsigned short* Kh  = (unsigned short*)(ws + off); off += HSZ * 2;
    unsigned short* VhT = (unsigned short*)(ws + off); off += HSZ * 2;
    float*          Obuf = (float*)(ws + off);         off += (size_t)BATCH * N_SEQ * C_DIM * 4;

    prep_weights_kernel<<<1024, 256, 0, stream>>>(Wq, Wk, Wv, Wo, WqT, WkT, WvT, WoT);

    dim3 gproj(256, 3);
    qkv_proj_kernel<<<gproj, 256, 0, stream>>>(q, kv, pos_q, pos_k,
                                               WqT, WkT, WvT, Qh, Kh, VhT);

    flash_attn_kernel<<<256, 256, 0, stream>>>(Qh, Kh, VhT, Obuf);

    out_proj_kernel<<<256, 256, 0, stream>>>(Obuf, WoT, bo, out);
}
